// QuantLinear_79224966742223
// MI455X (gfx1250) — compile-verified, matrix operand order checked by reading, not run on hardware
//
#include <hip/hip_runtime.h>

typedef __attribute__((ext_vector_type(16))) _Float16 v16h;
typedef __attribute__((ext_vector_type(8)))  _Float16 v8h;
typedef __attribute__((ext_vector_type(8)))  float    v8f;
typedef __attribute__((ext_vector_type(4)))  float    v4f;

#define F4_MAX   6.0f
#define F8_MAX   448.0f
#define GS_NUM   (F8_MAX * F4_MAX)   /* 2688 = global-scale numerator */

#define TILE_M   128
#define TILE_N   128
#define TILE_K   32
#define LDS_PITCH 40                 /* 32 halfs + 8 pad to spread LDS banks */

// ---------- quantization helpers (match the reference's RNE math) ----------

__device__ __forceinline__ float fp8_e4m3_rne_pos(float x) {
    // x >= 0. e4m3fn: 3 mantissa bits, min normal 2^-6, max 448, RNE.
    if (!(x > 0.0f)) return 0.0f;
    x = fminf(x, F8_MAX);
    int e = (int)floorf(log2f(x));
    e = e < -6 ? -6 : e;             // denormal floor: step 2^-9
    float step = exp2f((float)(e - 3));
    float q = rintf(x / step) * step; // rintf = round-half-to-even
    return fminf(q, F8_MAX);
}

__device__ __forceinline__ float fp4_e2m1_rne(float v) {
    float s = v < 0.0f ? -1.0f : 1.0f;
    float a = fminf(fabsf(v), F4_MAX);
    float e = fminf(fmaxf(floorf(log2f(fmaxf(a, 1.0f))), 0.0f), 2.0f);
    float step = exp2f(e - 1.0f);     // 0.5, 1.0 or 2.0
    return s * fminf(rintf(a / step) * step, F4_MAX);
}

// ---------- kernel 0: init amax slots ----------

__global__ void qlin_init_ws(unsigned* amax_bits) {
    if (threadIdx.x < 2) amax_bits[threadIdx.x] = 0u;
}

// ---------- kernel 1: |amax| reduction (float bits monotonic for >=0) ----------

__global__ __launch_bounds__(256)
void qlin_amax(const float* __restrict__ p, long long n, unsigned* __restrict__ out) {
    __shared__ float red[256];
    float m = 0.0f;
    for (long long i = (long long)blockIdx.x * blockDim.x + threadIdx.x; i < n;
         i += (long long)gridDim.x * blockDim.x)
        m = fmaxf(m, fabsf(p[i]));
    red[threadIdx.x] = m;
    __syncthreads();
    for (int s = 128; s > 0; s >>= 1) {
        if ((int)threadIdx.x < s) red[threadIdx.x] = fmaxf(red[threadIdx.x], red[threadIdx.x + s]);
        __syncthreads();
    }
    if (threadIdx.x == 0) atomicMax(out, __float_as_uint(red[0]));
}

// ---------- kernel 2: nvfp4 quant->dequant into exact fp16 (q*scale) ----------
// One thread per 16-element block along K. Stored value = q * e4m3_scale,
// which is exact in fp16 (<=6 significand bits, magnitude <= 2688).

__global__ __launch_bounds__(256)
void qlin_quantdq(const float* __restrict__ in, _Float16* __restrict__ outh,
                  long long nblocks, const unsigned* __restrict__ amax_bits) {
    long long b = (long long)blockIdx.x * blockDim.x + threadIdx.x;
    if (b >= nblocks) return;
    float amax_g = fmaxf(__uint_as_float(*amax_bits), 1e-30f);
    float gs = GS_NUM / amax_g;

    const v4f* src = (const v4f*)(in + b * 16);
    float v[16];
    float am = 0.0f;
#pragma unroll
    for (int i = 0; i < 4; ++i) {
        v4f t = src[i];
#pragma unroll
        for (int j = 0; j < 4; ++j) {
            v[i * 4 + j] = t[j];
            am = fmaxf(am, fabsf(t[j]));
        }
    }
    float scale = fp8_e4m3_rne_pos(am * (gs / F4_MAX));
    float safe  = scale > 0.0f ? scale : 1.0f;
    float inv   = gs / safe;

    v8h h0, h1;
#pragma unroll
    for (int i = 0; i < 8; ++i)
        h0[i] = (_Float16)(fp4_e2m1_rne(v[i] * inv) * safe);
#pragma unroll
    for (int i = 0; i < 8; ++i)
        h1[i] = (_Float16)(fp4_e2m1_rne(v[8 + i] * inv) * safe);

    v8h* dst = (v8h*)(outh + b * 16);
    dst[0] = h0;
    dst[1] = h1;
}

// ---------- kernel 3: tiled fp16 WMMA GEMM, double-buffered LDS ----------
// Block: 256 threads = 8 waves, 128x128 tile. Waves: 4 along M x 2 along N,
// each wave owns 32x64 = 2x4 tiles of 16x16 (8 x v8f accumulators).
// Tile s+1 is staged into LDS buffer (s+1)&1 while tile s is consumed from
// buffer s&1; a single workgroup barrier per K-step orders producer/consumer
// (the WAR on a buffer is two iterations apart, covered by that barrier).

__global__ __launch_bounds__(256)
void qlin_gemm(const _Float16* __restrict__ Aq,  // [M,K] fp16 (q*scale)
               const _Float16* __restrict__ Bq,  // [N,K] fp16 (q*scale)
               const float* __restrict__ bias,
               float* __restrict__ out,          // [M,N]
               int M, int N, int K,
               const unsigned* __restrict__ amax_bits) {
    __shared__ _Float16 lA[2][TILE_M * LDS_PITCH];
    __shared__ _Float16 lB[2][TILE_N * LDS_PITCH];

    const int tid   = threadIdx.x;
    const int lane  = tid & 31;
    const int wid   = tid >> 5;
    const int waveM = wid & 3;   // 0..3
    const int waveN = wid >> 2;  // 0..1
    const int wgN   = blockIdx.x * TILE_N;
    const int wgM   = blockIdx.y * TILE_M;

    const v8f vzero = {0.f, 0.f, 0.f, 0.f, 0.f, 0.f, 0.f, 0.f};
    v8f acc[2][4];
#pragma unroll
    for (int mt = 0; mt < 2; ++mt)
#pragma unroll
        for (int nt = 0; nt < 4; ++nt) acc[mt][nt] = vzero;

    // ISA 16-bit operand layout: lane%16 = row(M) / col(N); lanes 16..31 take
    // K halves offset by +8; VGPRs 0..3 = K[0..7|8..15], VGPRs 4..7 = +16.
    const int row16    = lane & 15;
    const int laneHalf = (lane & 16) ? 8 : 0;

    union Frag { v16h v; struct { v8h lo; v8h hi; } p; };

    // Cooperative stage: 512 chunks of 8 halfs per tile, 2 chunks/thread.
    auto stage = [&](int buf, int k0) {
        const int c0 = tid * 2;
#pragma unroll
        for (int c = c0; c < c0 + 2; ++c) {
            const int r  = c >> 2;
            const int kc = (c & 3) * 8;
            *(v8h*)&lA[buf][r * LDS_PITCH + kc] =
                *(const v8h*)&Aq[(long long)(wgM + r) * K + k0 + kc];
            *(v8h*)&lB[buf][r * LDS_PITCH + kc] =
                *(const v8h*)&Bq[(long long)(wgN + r) * K + k0 + kc];
        }
    };

    stage(0, 0);
    __syncthreads();

    const int nsteps = K / TILE_K;
    for (int s = 0; s < nsteps; ++s) {
        const int k0  = s * TILE_K;
        const int cur = s & 1;

        // Producer for the next tile: overlaps with this tile's WMMAs.
        if (s + 1 < nsteps) stage(cur ^ 1, k0 + TILE_K);
        if (s + 2 < nsteps) {  // near-scope prefetch of tile s+2 (locality=3)
            __builtin_prefetch(&Aq[(long long)(wgM + (tid >> 1)) * K + k0 + 2 * TILE_K], 0, 3);
            __builtin_prefetch(&Bq[(long long)(wgN + (tid >> 1)) * K + k0 + 2 * TILE_K], 0, 3);
        }

        Frag a[2], b[4];
#pragma unroll
        for (int mt = 0; mt < 2; ++mt) {
            const int r = waveM * 32 + mt * 16 + row16;
            a[mt].p.lo = *(const v8h*)&lA[cur][r * LDS_PITCH + laneHalf];
            a[mt].p.hi = *(const v8h*)&lA[cur][r * LDS_PITCH + 16 + laneHalf];
        }
#pragma unroll
        for (int nt = 0; nt < 4; ++nt) {
            const int r = waveN * 64 + nt * 16 + row16;
            b[nt].p.lo = *(const v8h*)&lB[cur][r * LDS_PITCH + laneHalf];
            b[nt].p.hi = *(const v8h*)&lB[cur][r * LDS_PITCH + 16 + laneHalf];
        }

#pragma unroll
        for (int mt = 0; mt < 2; ++mt)
#pragma unroll
            for (int nt = 0; nt < 4; ++nt)
                acc[mt][nt] = __builtin_amdgcn_wmma_f32_16x16x32_f16(
                    /*neg_a=*/false, a[mt].v, /*neg_b=*/false, b[nt].v,
                    /*c_mod=*/(short)0, acc[mt][nt],
                    /*reuse_a=*/false, /*reuse_b=*/false);

        __syncthreads();  // next-tile writes visible; frees buffer cur for s+2
    }

    // Epilogue: alpha = 1/(gs_x*gs_w) = amax_x*amax_w / 2688^2, plus bias.
    const float ax    = fmaxf(__uint_as_float(amax_bits[0]), 1e-30f);
    const float aw    = fmaxf(__uint_as_float(amax_bits[1]), 1e-30f);
    const float alpha = (ax * aw) / (GS_NUM * GS_NUM);

    const int colLane = lane & 15;
    const int rowOff  = (lane >> 4) * 8;  // D layout: lanes 16..31 -> M+8
#pragma unroll
    for (int mt = 0; mt < 2; ++mt)
#pragma unroll
        for (int nt = 0; nt < 4; ++nt) {
            const int col = wgN + waveN * 64 + nt * 16 + colLane;
            const float bv = bias[col];
            const int mBase = wgM + waveM * 32 + mt * 16 + rowOff;
#pragma unroll
            for (int r = 0; r < 8; ++r)
                out[(long long)(mBase + r) * N + col] = acc[mt][nt][r] * alpha + bv;
        }
}

// ---------- host launcher ----------

extern "C" void kernel_launch(void* const* d_in, const int* in_sizes, int n_in,
                              void* d_out, int out_size, void* d_ws, size_t ws_size,
                              hipStream_t stream) {
    const float* x    = (const float*)d_in[0];  // [B,S,K]
    const float* W    = (const float*)d_in[1];  // [N,K]
    const float* bias = (const float*)d_in[2];  // [N]

    const long long nx = in_sizes[0];
    const long long nw = in_sizes[1];
    const int N = in_sizes[2];
    const int K = (int)(nw / N);
    const int M = (int)(nx / K);

    // Workspace layout: [2 x u32 amax bits][pad to 256B][xq fp16][wq fp16]
    unsigned* amax_bits = (unsigned*)d_ws;
    _Float16* xq = (_Float16*)((char*)d_ws + 256);
    _Float16* wq = xq + nx;

    qlin_init_ws<<<1, 64, 0, stream>>>(amax_bits);

    qlin_amax<<<1024, 256, 0, stream>>>(x, nx, amax_bits + 0);
    qlin_amax<<<1024, 256, 0, stream>>>(W, nw, amax_bits + 1);

    const long long nbx = nx / 16, nbw = nw / 16;
    qlin_quantdq<<<(unsigned)((nbx + 255) / 256), 256, 0, stream>>>(x, xq, nbx, amax_bits + 0);
    qlin_quantdq<<<(unsigned)((nbw + 255) / 256), 256, 0, stream>>>(W, wq, nbw, amax_bits + 1);

    dim3 grid(N / TILE_N, M / TILE_M);
    qlin_gemm<<<grid, 256, 0, stream>>>(xq, wq, bias, (float*)d_out, M, N, K, amax_bits);
}
